// HE_graph_hypergraph_surv_83494164234284
// MI455X (gfx1250) — matrix operand order repaired and checked by reading, not compiled, not run on hardware
//
#include <hip/hip_runtime.h>

#define N_NODES  100000
#define N_EDGES  800000
#define N_HEDGES 25000
#define BG       16
#define F_INDIM  64
#define HDIM     128

typedef __attribute__((ext_vector_type(16))) _Float16 v16h;
typedef __attribute__((ext_vector_type(8)))  float    v8f;

// CDNA5 packed-f16 global atomic add (no return -> STOREcnt), §15.18 op 89.
__device__ __forceinline__ void pk_atomic_add_f16(_Float16* addr, unsigned int packed)
{
  asm volatile("global_atomic_pk_add_f16 %0, %1, off"
               :
               : "v"((unsigned long long)(size_t)addr), "v"(packed)
               : "memory");
}

// ---------------------------------------------------------------------------
// WMMA GEMM: C[rows x 128] = act(A[rows x K] @ B[K x 128] (+ bias))
// 256 threads = 8 wave32; wave w owns cols [16w,16w+16). A tile staged in LDS
// as packed f16 pairs: the ISA A-fragment (16-bit A 16x32) keeps K-adjacent
// pairs in one dword, so each fragment dword is ONE aligned LDS u32 read.
// ---------------------------------------------------------------------------
template <int K, bool A_F16, bool OUT_F16, bool RELU, bool BIAS>
__global__ __launch_bounds__(256) void gemm_wmma_kernel(
    const void* __restrict__ Avoid, const float* __restrict__ B,
    const float* __restrict__ bias, void* __restrict__ Cvoid)
{
  const int KP = K / 2;                     // u32 pairs per row
  __shared__ unsigned int As[16 * (K / 2)];
  const int lane = threadIdx.x & 31;
  const int wave = threadIdx.x >> 5;
  const long rowBase = (long)blockIdx.x * 16;

  if (A_F16) {
    const unsigned int* Ain = (const unsigned int*)Avoid;
    for (int i = threadIdx.x; i < 16 * KP; i += 256) {
      int r = i / KP, c = i - r * KP;
      As[i] = Ain[(rowBase + r) * KP + c];
    }
  } else {
    const float* Af = (const float*)Avoid;
    for (int i = threadIdx.x; i < 16 * KP; i += 256) {
      int r = i / KP, c = i - r * KP;
      float2 f = ((const float2*)(Af + (rowBase + r) * K))[c];
      union { _Float16 h[2]; unsigned int u; } pk;
      pk.h[0] = (_Float16)f.x; pk.h[1] = (_Float16)f.y;
      As[i] = pk.u;
    }
  }
  __syncthreads();

  const int colBase = wave * 16;
  const int m  = lane & 15;     // A row / B col / C col for this lane
  const int hi = lane >> 4;     // lane half

  v8f acc = {0.f, 0.f, 0.f, 0.f, 0.f, 0.f, 0.f, 0.f};
  #pragma unroll
  for (int k0 = 0; k0 < K; k0 += 32) {
    union { v16h h; unsigned int u[8]; } a;
    // A fragment: lanes 0-15 K-groups {0..7,16..23}, lanes 16-31 {8..15,24..31}
    const int kgA = hi * 8;
    #pragma unroll
    for (int v = 0; v < 8; ++v) {
      int kk = (v < 4) ? (kgA + 2 * v) : (16 + kgA + 2 * (v - 4));
      a.u[v] = As[m * KP + ((k0 + kk) >> 1)];     // packed pair (k, k+1)
    }
    // B fragment: lanes 0-15 hold K {0..7,16..23}, lanes 16-31 {8..15,24..31} for col n
    v16h b;
    const int kgB = hi * 8;
    const float* Bp = B + (long)k0 * HDIM + colBase + m;
    #pragma unroll
    for (int v = 0; v < 8; ++v) {
      int kk = (v < 4) ? (kgB + 2 * v) : (16 + kgB + 2 * (v - 4));
      b[2 * v]     = (_Float16)Bp[(long)kk * HDIM];
      b[2 * v + 1] = (_Float16)Bp[(long)(kk + 1) * HDIM];
    }
    acc = __builtin_amdgcn_wmma_f32_16x16x32_f16(false, a.h, false, b,
                                                 (short)0, acc, false, false);
  }

  // C/D layout: VGPR r -> row (r + 8*hi), col = m
  const int mh = hi * 8;
  #pragma unroll
  for (int r = 0; r < 8; ++r) {
    float v = acc[r];
    if (BIAS) v += bias[colBase + m];
    if (RELU) v = fmaxf(v, 0.f);
    long idx = (rowBase + mh + r) * HDIM + colBase + m;
    if (OUT_F16) ((_Float16*)Cvoid)[idx] = (_Float16)v;
    else         ((float*)Cvoid)[idx]    = v;
  }
}

// ---------------------------------------------------------------------------
// Graph structure degrees (once per launch)
// ---------------------------------------------------------------------------
__global__ void degree_kernel(const int* __restrict__ ni, const int* __restrict__ he,
                              float* __restrict__ deg, float* __restrict__ cnt)
{
  int e = blockIdx.x * blockDim.x + threadIdx.x;
  if (e < N_EDGES) {
    atomicAdd(&deg[ni[e]], 1.0f);
    atomicAdd(&cnt[he[e]], 1.0f);
  }
}

// edge[he[e]] += t[ni[e]]  (one wave per incidence; lane: 4 halfs = uint2)
__global__ void scatter_edges_kernel(const _Float16* __restrict__ t,
                                     const int* __restrict__ ni,
                                     const int* __restrict__ he,
                                     _Float16* __restrict__ edge)
{
  int tid = blockIdx.x * blockDim.x + threadIdx.x;
  int e = tid >> 5, lane = tid & 31;
  if (e >= N_EDGES) return;
  const uint2 v = ((const uint2*)(t + (size_t)ni[e] * HDIM))[lane];
  _Float16* dst = edge + (size_t)he[e] * HDIM + lane * 4;
  pk_atomic_add_f16(dst,     v.x);
  pk_atomic_add_f16(dst + 2, v.y);
}

// edge *= 1/|e|   (one thread per packed pair)
__global__ void scale_edges_kernel(unsigned int* __restrict__ edge,
                                   const float* __restrict__ cnt)
{
  int i = blockIdx.x * blockDim.x + threadIdx.x;
  if (i < N_HEDGES * (HDIM / 2)) {
    float c = cnt[i >> 6];
    float s = (c > 0.f) ? (1.f / c) : 0.f;
    union { _Float16 h[2]; unsigned int u; } pk;
    pk.u = edge[i];
    pk.h[0] = (_Float16)((float)pk.h[0] * s);
    pk.h[1] = (_Float16)((float)pk.h[1] * s);
    edge[i] = pk.u;
  }
}

// acc[ni[e]] += edge[he[e]]
__global__ void gather_nodes_kernel(const _Float16* __restrict__ edge,
                                    const int* __restrict__ ni,
                                    const int* __restrict__ he,
                                    _Float16* __restrict__ acc)
{
  int tid = blockIdx.x * blockDim.x + threadIdx.x;
  int e = tid >> 5, lane = tid & 31;
  if (e >= N_EDGES) return;
  const uint2 v = ((const uint2*)(edge + (size_t)he[e] * HDIM))[lane];
  _Float16* dst = acc + (size_t)ni[e] * HDIM + lane * 4;
  pk_atomic_add_f16(dst,     v.x);
  pk_atomic_add_f16(dst + 2, v.y);
}

// acc = relu(acc * d_inv + bias), in f16
__global__ void finalize_nodes_kernel(_Float16* __restrict__ acc,
                                      const float* __restrict__ deg,
                                      const float* __restrict__ bias)
{
  int i = blockIdx.x * blockDim.x + threadIdx.x;
  if (i < N_NODES * HDIM) {
    float d = deg[i >> 7];
    float dinv = (d > 0.f) ? (1.f / d) : 0.f;
    float v = (float)acc[i] * dinv + bias[i & (HDIM - 1)];
    acc[i] = (_Float16)fmaxf(v, 0.f);
  }
}

// f32 pooled sums from f16 activations
__global__ void pool_kernel(const _Float16* __restrict__ h, const int* __restrict__ batch,
                            float* __restrict__ psum, float* __restrict__ pcnt)
{
  int tid = blockIdx.x * blockDim.x + threadIdx.x;
  int n = tid >> 5, lane = tid & 31;
  if (n >= N_NODES) return;
  int b = batch[n];
  const uint2 v = ((const uint2*)(h + (size_t)n * HDIM))[lane];
  union { _Float16 h2[2]; unsigned int u; } a, c;
  a.u = v.x; c.u = v.y;
  float* dst = psum + b * HDIM + lane * 4;
  atomicAdd(dst + 0, (float)a.h2[0]); atomicAdd(dst + 1, (float)a.h2[1]);
  atomicAdd(dst + 2, (float)c.h2[0]); atomicAdd(dst + 3, (float)c.h2[1]);
  if (pcnt && lane == 0) atomicAdd(&pcnt[b], 1.0f);
}

// ---------------------------------------------------------------------------
// MLP head: 16 rows only — single block, LDS staged, f32
// ---------------------------------------------------------------------------
__global__ __launch_bounds__(128) void head_kernel(
    const float* __restrict__ p1s, const float* __restrict__ p2s,
    const float* __restrict__ pcnt,
    const float* __restrict__ Wg, const float* __restrict__ bg,
    const float* __restrict__ W1, const float* __restrict__ b1,
    const float* __restrict__ g1, const float* __restrict__ be1,
    const float* __restrict__ rm1, const float* __restrict__ rv1,
    const float* __restrict__ W2, const float* __restrict__ b2,
    const float* __restrict__ g2, const float* __restrict__ be2,
    const float* __restrict__ rm2, const float* __restrict__ rv2,
    const float* __restrict__ W3, float* __restrict__ out)
{
  __shared__ float cat[BG][2 * HDIM];
  __shared__ float gb[BG][HDIM];
  __shared__ float h1[BG][64];
  __shared__ float h2[BG][32];
  const int t = threadIdx.x;

  for (int i = t; i < BG * HDIM; i += 128) {
    int b = i >> 7, f = i & 127;
    float c = fmaxf(pcnt[b], 1.0f);
    cat[b][f]        = p1s[b * HDIM + f] / c;
    cat[b][HDIM + f] = p2s[b * HDIM + f] / c;
  }
  __syncthreads();

  {  // g = cat @ Wg + bg   (256 -> 128)
    float s[BG];
    #pragma unroll
    for (int b = 0; b < BG; ++b) s[b] = bg[t];
    for (int k = 0; k < 2 * HDIM; ++k) {
      float w = Wg[k * HDIM + t];
      #pragma unroll
      for (int b = 0; b < BG; ++b) s[b] += cat[b][k] * w;
    }
    #pragma unroll
    for (int b = 0; b < BG; ++b) gb[b][t] = s[b];
  }
  __syncthreads();

  if (t < 64) {  // h1 = relu(bn(g @ W1 + b1))
    float sc = g1[t] * rsqrtf(rv1[t] + 1e-5f);
    for (int b = 0; b < BG; ++b) {
      float s = b1[t];
      for (int k = 0; k < HDIM; ++k) s += gb[b][k] * W1[k * 64 + t];
      h1[b][t] = fmaxf((s - rm1[t]) * sc + be1[t], 0.f);
    }
  }
  __syncthreads();

  if (t < 32) {  // h2 = relu(bn(h1 @ W2 + b2))
    float sc = g2[t] * rsqrtf(rv2[t] + 1e-5f);
    for (int b = 0; b < BG; ++b) {
      float s = b2[t];
      for (int k = 0; k < 64; ++k) s += h1[b][k] * W2[k * 32 + t];
      h2[b][t] = fmaxf((s - rm2[t]) * sc + be2[t], 0.f);
    }
  }
  __syncthreads();

  if (t < 4) {  // out = h2 @ W3
    for (int b = 0; b < BG; ++b) {
      float s = 0.f;
      for (int k = 0; k < 32; ++k) s += h2[b][k] * W3[k * 4 + t];
      out[b * 4 + t] = s;
    }
  }
}

// ---------------------------------------------------------------------------
extern "C" void kernel_launch(void* const* d_in, const int* in_sizes, int n_in,
                              void* d_out, int out_size, void* d_ws, size_t ws_size,
                              hipStream_t stream)
{
  (void)in_sizes; (void)n_in; (void)out_size; (void)ws_size;
  const float* x         = (const float*)d_in[0];
  const int*   node_idx  = (const int*)d_in[1];
  const int*   hedge_idx = (const int*)d_in[2];
  const int*   batch     = (const int*)d_in[3];
  const float* W0  = (const float*)d_in[4];  const float* b0  = (const float*)d_in[5];
  const float* Wc1 = (const float*)d_in[6];  const float* bc1 = (const float*)d_in[7];
  const float* Wc2 = (const float*)d_in[8];  const float* bc2 = (const float*)d_in[9];
  const float* Wg  = (const float*)d_in[10]; const float* bg  = (const float*)d_in[11];
  const float* W1  = (const float*)d_in[12]; const float* b1  = (const float*)d_in[13];
  const float* g1  = (const float*)d_in[14]; const float* be1 = (const float*)d_in[15];
  const float* rm1 = (const float*)d_in[16]; const float* rv1 = (const float*)d_in[17];
  const float* W2  = (const float*)d_in[18]; const float* b2  = (const float*)d_in[19];
  const float* g2  = (const float*)d_in[20]; const float* be2 = (const float*)d_in[21];
  const float* rm2 = (const float*)d_in[22]; const float* rv2 = (const float*)d_in[23];
  const float* W3  = (const float*)d_in[24];
  float* out = (float*)d_out;

  // Workspace carve-out: f16 activation buffers first, then f32 scalars.
  const size_t NH = (size_t)N_NODES * HDIM;
  const size_t MH = (size_t)N_HEDGES * HDIM;
  char* base = (char*)d_ws;
  _Float16* h0a  = (_Float16*)base;                 // N x 128 f16 (first_h / conv2 acc)
  _Float16* tb   = h0a + NH;                        // N x 128 f16 (h @ Wc)
  _Float16* hc   = tb + NH;                         // N x 128 f16 (conv1 out)
  _Float16* edge = hc + NH;                         // M x 128 f16
  float*    deg  = (float*)(edge + MH);             // N (cnt follows: joint memset)
  float*    cnt  = deg + N_NODES;                   // M
  float*    p1   = cnt + N_HEDGES;                  // 16 x 128 (p1,p2,pc contiguous)
  float*    p2   = p1 + BG * HDIM;
  float*    pc   = p2 + BG * HDIM;

  const int rowTiles = N_NODES / 16;                // 6250, exact
  const int eThreads = N_EDGES * 32;
  const int nThreads = N_NODES * 32;
  const int edgePairs = N_HEDGES * (HDIM / 2);

  // degrees / hyperedge sizes
  hipMemsetAsync(deg, 0, (N_NODES + N_HEDGES) * sizeof(float), stream);
  degree_kernel<<<(N_EDGES + 255) / 256, 256, 0, stream>>>(node_idx, hedge_idx, deg, cnt);

  // h0 = relu(x @ W0 + b0)  -> f16
  gemm_wmma_kernel<F_INDIM, false, true, true, true>
      <<<rowTiles, 256, 0, stream>>>(x, W0, b0, h0a);

  // ---- conv1 ----
  gemm_wmma_kernel<HDIM, true, true, false, false>
      <<<rowTiles, 256, 0, stream>>>(h0a, Wc1, nullptr, tb);
  hipMemsetAsync(edge, 0, MH * sizeof(_Float16), stream);
  scatter_edges_kernel<<<eThreads / 256, 256, 0, stream>>>(tb, node_idx, hedge_idx, edge);
  scale_edges_kernel<<<(edgePairs + 255) / 256, 256, 0, stream>>>((unsigned int*)edge, cnt);
  hipMemsetAsync(hc, 0, NH * sizeof(_Float16), stream);
  gather_nodes_kernel<<<eThreads / 256, 256, 0, stream>>>(edge, node_idx, hedge_idx, hc);
  finalize_nodes_kernel<<<(int)((NH + 255) / 256), 256, 0, stream>>>(hc, deg, bc1);

  // pool 1 (+ per-graph node counts)
  hipMemsetAsync(p1, 0, (2 * BG * HDIM + BG) * sizeof(float), stream);
  pool_kernel<<<(nThreads + 255) / 256, 256, 0, stream>>>(hc, batch, p1, pc);

  // ---- conv2 (reuse h0a as accumulator/output) ----
  gemm_wmma_kernel<HDIM, true, true, false, false>
      <<<rowTiles, 256, 0, stream>>>(hc, Wc2, nullptr, tb);
  hipMemsetAsync(edge, 0, MH * sizeof(_Float16), stream);
  scatter_edges_kernel<<<eThreads / 256, 256, 0, stream>>>(tb, node_idx, hedge_idx, edge);
  scale_edges_kernel<<<(edgePairs + 255) / 256, 256, 0, stream>>>((unsigned int*)edge, cnt);
  hipMemsetAsync(h0a, 0, NH * sizeof(_Float16), stream);
  gather_nodes_kernel<<<eThreads / 256, 256, 0, stream>>>(edge, node_idx, hedge_idx, h0a);
  finalize_nodes_kernel<<<(int)((NH + 255) / 256), 256, 0, stream>>>(h0a, deg, bc2);

  // pool 2 (counts already computed)
  pool_kernel<<<(nThreads + 255) / 256, 256, 0, stream>>>(h0a, batch, p2, nullptr);

  // MLP head on 16 graphs
  head_kernel<<<1, 128, 0, stream>>>(p1, p2, pc, Wg, bg, W1, b1, g1, be1, rm1, rv1,
                                     W2, b2, g2, be2, rm2, rv2, W3, out);
}
